// Attention_8100308321041
// MI455X (gfx1250) — matrix operand order repaired and checked by reading, not compiled, not run on hardware
//
#include <hip/hip_runtime.h>
#include <hip/hip_bf16.h>

typedef __attribute__((ext_vector_type(16))) _Float16 v16h;
typedef __attribute__((ext_vector_type(8)))  _Float16 v8h;
typedef __attribute__((ext_vector_type(8)))  float    v8f;

#define HD 16     // head dim
#define NT 256    // tokens per window
#define NH 8      // heads
#define NWIN 64   // windows per image

// ---------------------------------------------------------------------------
// Kernel 1: resolve relative-position-bias gather once:
//   bias[h][i*N+j] = table[index[i*N+j]][h]
// ---------------------------------------------------------------------------
__global__ __launch_bounds__(256)
void bias_gather_kernel(const long long* __restrict__ index,
                        const float* __restrict__ table,
                        float* __restrict__ bias) {
  const int ij = blockIdx.x * 256 + threadIdx.x;   // 0 .. 65535
  const long long idx = index[ij];
  const float* trow = table + (size_t)idx * NH;
  #pragma unroll
  for (int hh = 0; hh < NH; ++hh)
    bias[(size_t)hh * (NT * NT) + ij] = trow[hh];
}

// ---------------------------------------------------------------------------
// Kernel 2: fused window attention for one (b,h) per workgroup.
//   128 threads = 4 waves; each wave handles 4 row-blocks of 16 rows.
// ---------------------------------------------------------------------------
__global__ __launch_bounds__(128)
void swin_attn_kernel(const float* __restrict__ q,
                      const float* __restrict__ k,
                      const float* __restrict__ v,
                      const float* __restrict__ biasT,  // [H][N][N]
                      const float* __restrict__ mask,   // [NWIN][N][N]
                      float* __restrict__ out) {        // [B][N][H*HD]
  __shared__ __align__(32) _Float16 Qh[NT * HD];       // 8 KB, row-major f16 (L2-normalized)
  __shared__ __align__(32) _Float16 Kh[NT * HD];       // 8 KB
  __shared__ __align__(32) _Float16 Vt[HD * NT];       // 8 KB, transposed [d][token]
  __shared__ __align__(32) _Float16 Pst[4][NT * HD];   // 32 KB, per-wave P staging [16][256]
  __shared__ __align__(32) _Float16 Zpad[32];          // 64 B of zeros for K-dim padding

  const int tid = threadIdx.x;
  const int bh  = blockIdx.x;          // b*8 + h
  const int hgl = bh & 7;              // head
  const int bgl = bh >> 3;             // window-batch index
  const int w   = bgl & (NWIN - 1);    // window id within image

  const float* qb = q + (size_t)bh * (NT * HD);
  const float* kb = k + (size_t)bh * (NT * HD);
  const float* vb = v + (size_t)bh * (NT * HD);

  if (tid < 32) Zpad[tid] = (_Float16)0.f;

  // ---- Phase 1: load + L2-normalize Q,K -> f16 LDS; V -> transposed f16 LDS
  for (int t = tid; t < NT; t += 128) {
    float buf[16];
    {
      const float4* p = (const float4*)(qb + (size_t)t * HD);
      ((float4*)buf)[0] = p[0]; ((float4*)buf)[1] = p[1];
      ((float4*)buf)[2] = p[2]; ((float4*)buf)[3] = p[3];
      float ss = 0.f;
      #pragma unroll
      for (int j = 0; j < 16; ++j) ss += buf[j] * buf[j];
      const float sc = 1.0f / fmaxf(sqrtf(ss), 1e-12f);
      v16h hv;
      #pragma unroll
      for (int j = 0; j < 16; ++j) hv[j] = (_Float16)(buf[j] * sc);
      *(v16h*)&Qh[t * HD] = hv;
    }
    {
      const float4* p = (const float4*)(kb + (size_t)t * HD);
      ((float4*)buf)[0] = p[0]; ((float4*)buf)[1] = p[1];
      ((float4*)buf)[2] = p[2]; ((float4*)buf)[3] = p[3];
      float ss = 0.f;
      #pragma unroll
      for (int j = 0; j < 16; ++j) ss += buf[j] * buf[j];
      const float sc = 1.0f / fmaxf(sqrtf(ss), 1e-12f);
      v16h hv;
      #pragma unroll
      for (int j = 0; j < 16; ++j) hv[j] = (_Float16)(buf[j] * sc);
      *(v16h*)&Kh[t * HD] = hv;
    }
    {
      const float4* p = (const float4*)(vb + (size_t)t * HD);
      ((float4*)buf)[0] = p[0]; ((float4*)buf)[1] = p[1];
      ((float4*)buf)[2] = p[2]; ((float4*)buf)[3] = p[3];
      #pragma unroll
      for (int j = 0; j < 16; ++j) Vt[j * NT + t] = (_Float16)buf[j];
    }
  }
  __syncthreads();

  // ---- Phase 2: per-wave attention over row-blocks
  const int lane = tid & 31;
  const int wv   = tid >> 5;       // wave 0..3
  const int hl   = lane >> 4;      // half-wave: 0 or 1
  const int lr   = lane & 15;      // lane-in-half

  const float* biasH = biasT + (size_t)hgl * (NT * NT);
  const float* maskW = mask  + (size_t)w   * (NT * NT);
  _Float16* Pw = &Pst[wv][0];

  // B-fragment source for QK^T: lanes 0-15 read real K rows (K=0..15 = d),
  // lanes 16-31 read the zero pad (K=16..31 padding) with step 0.
  // Address-select instead of data-select: no per-element cndmasks.
  const _Float16* kfrag0 = hl ? (const _Float16*)Zpad : &Kh[lr * HD];
  const int       kstep  = hl ? 0 : (16 * HD);   // halves per column tile

  const v8f zeroC = {};

  for (int rr = 0; rr < 4; ++rr) {
    const int rb = wv * 4 + rr;    // row-block 0..15
    const int row0 = rb * 16 + hl * 8;           // first C-layout row this lane touches

    // prefetch next row-block's bias/mask rows into near caches
    if (rr < 3) {
      __builtin_prefetch(biasH + (size_t)(row0 + 16) * NT + lr, 0, 1);
      __builtin_prefetch(maskW + (size_t)(row0 + 16) * NT + lr, 0, 1);
    }

    // A fragment (Q rows, K-dim = 16 padded to 32): lanes 0-15 elems 0..15 = d 0..15,
    // lanes 16-31 correspond to K>=16 -> zero.
    v16h a = {};
    if (!hl) a = *(const v16h*)&Qh[(rb * 16 + lr) * HD];

    // ---- S = Qn * Kn^T + (bias + mask) : C operand carries the additive terms
    v8f s[16];
    const _Float16* kptr = kfrag0;
    #pragma unroll
    for (int jt = 0; jt < 16; ++jt) {
      const int col = jt * 16 + lr;
      v8f cin;
      #pragma unroll
      for (int i = 0; i < 8; ++i) {
        const size_t o = (size_t)(row0 + i) * NT + col;
        cin[i] = biasH[o] + maskW[o];
      }
      v16h kv = *(const v16h*)kptr;
      kptr += kstep;
      s[jt] = __builtin_amdgcn_wmma_f32_16x16x32_f16(
          false, a, false, kv, (short)0, cin, false, false);
    }

    // ---- softmax across the 256 columns (per row)
    float mx[8], sminv[8];
    #pragma unroll
    for (int i = 0; i < 8; ++i) {
      float m = s[0][i];
      #pragma unroll
      for (int jt = 1; jt < 16; ++jt) m = fmaxf(m, s[jt][i]);
      #pragma unroll
      for (int off = 8; off >= 1; off >>= 1)
        m = fmaxf(m, __shfl_xor(m, off, 32));          // stays within 16-lane half
      mx[i] = m;
    }
    #pragma unroll
    for (int i = 0; i < 8; ++i) {
      float acc = 0.f;
      #pragma unroll
      for (int jt = 0; jt < 16; ++jt) {
        const float e = __expf(s[jt][i] - mx[i]);
        s[jt][i] = e;
        acc += e;
      }
      #pragma unroll
      for (int off = 8; off >= 1; off >>= 1)
        acc += __shfl_xor(acc, off, 32);
      sminv[i] = 1.0f / acc;                           // >= exp(0) = 1, never zero
    }

    // ---- stage P as f16 row-major [16][256] (C-layout -> A-layout via LDS)
    #pragma unroll
    for (int jt = 0; jt < 16; ++jt) {
      const int col = jt * 16 + lr;
      #pragma unroll
      for (int i = 0; i < 8; ++i)
        Pw[(hl * 8 + i) * NT + col] = (_Float16)(s[jt][i] * sminv[i]);
    }
    __syncthreads();

    // ---- O = P * V : 8 chained WMMAs (K=32 each over 256 tokens)
    v8f acc = zeroC;
    #pragma unroll
    for (int kt = 0; kt < 8; ++kt) {
      v16h pa = *(const v16h*)&Pw[lr * NT + kt * 32 + hl * 16];    // K = kt*32+hl*16+0..15
      // B fragment from V^T: column = d = lr, K tokens contiguous
      v16h bv = *(const v16h*)&Vt[lr * NT + kt * 32 + hl * 16];
      acc = __builtin_amdgcn_wmma_f32_16x16x32_f16(
          false, pa, false, bv, (short)0, acc, false, false);
    }

    // ---- store out[b][row][h*16 + d], coalesced across lanes
    float* ob = out + ((size_t)bgl * NT + (size_t)rb * 16) * (NH * HD) + hgl * HD;
    #pragma unroll
    for (int i = 0; i < 8; ++i)
      ob[(size_t)(hl * 8 + i) * (NH * HD) + lr] = acc[i];

    __syncthreads();
  }
}

// ---------------------------------------------------------------------------
extern "C" void kernel_launch(void* const* d_in, const int* in_sizes, int n_in,
                              void* d_out, int out_size, void* d_ws, size_t ws_size,
                              hipStream_t stream) {
  (void)in_sizes; (void)n_in; (void)out_size; (void)ws_size;
  const float*     q     = (const float*)d_in[0];
  const float*     k     = (const float*)d_in[1];
  const float*     v     = (const float*)d_in[2];
  const float*     table = (const float*)d_in[3];
  const long long* index = (const long long*)d_in[4];   // int64 per reference
  const float*     mask  = (const float*)d_in[5];
  float* out  = (float*)d_out;
  float* bias = (float*)d_ws;   // H*N*N floats = 2 MB scratch

  bias_gather_kernel<<<(NT * NT) / 256, 256, 0, stream>>>(index, table, bias);
  swin_attn_kernel<<<512 * NH, 128, 0, stream>>>(q, k, v, bias, mask, out);
}
